// GraphNeuralSolver_78847009620172
// MI455X (gfx1250) — matrix operand order repaired and checked by reading, not compiled, not run on hardware
//
#include <hip/hip_runtime.h>

typedef __attribute__((ext_vector_type(16))) __bf16 v16bf;
typedef __attribute__((ext_vector_type(8)))  float  v8f;

#define HDIM 128
#define KITER 4
#define GAMMA_F 0.9f

// LDS strides (elements). All chosen so row stride is a multiple of 8 elements
// (16B -> ds_load_b128 stays legal) and not 0 mod 64 dwords (bank spread).
#define NODE_STRIDE 296  // node fv tile, 288 live cols
#define MSG_STRIDE 168   // edge fv tile, 160 live cols
#define HID_STRIDE 132   // f32 hidden
#define HB_STRIDE 136    // bf16 hidden staging

// ---------- bf16 helpers ----------
__device__ inline __bf16 f2bf(float f) {
  unsigned u = __builtin_bit_cast(unsigned, f);
  unsigned r = (u + 0x7FFFu + ((u >> 16) & 1u)) >> 16;
  unsigned short s = (unsigned short)r;
  return __builtin_bit_cast(__bf16, s);
}

__device__ inline v8f wmma_bf16(v16bf a, v16bf b, v8f c) {
  return __builtin_amdgcn_wmma_f32_16x16x32_bf16(false, a, false, b, (short)0, c,
                                                 false, false);
}

// A-matrix 16x32 bf16 fragment from row-major bf16 (LDS). K-contiguous per lane:
// two 16B chunks per lane -> compiler emits 2x ds_load_b128.
__device__ inline v16bf load_a_bf16(const __bf16* A, int lda, int lane) {
  int lo = lane & 15, hi = lane >> 4;
  v16bf a;
#pragma unroll
  for (int j = 0; j < 8; ++j) {
    int k0 = 2 * j + hi * 16;
    a[2 * j]     = A[lo * lda + k0];
    a[2 * j + 1] = A[lo * lda + k0 + 1];
  }
  return a;
}

// B-matrix fragment from pre-swizzled weights: lane's 16 values are contiguous
// (32B) -> 2x global_load_b128, no repacking.
__device__ inline v16bf load_b_swz(const __bf16* Wswz, int kc, int ct, int lane) {
  const __bf16* p = Wswz + ((((size_t)kc * 8 + ct) * 32 + lane) << 4);
  return *(const v16bf*)p;
}

__device__ inline float grp8_sum(float x) {
  x += __shfl_xor(x, 1);
  x += __shfl_xor(x, 2);
  x += __shfl_xor(x, 4);
  return x;
}

// ---------- fused GEMM helpers (32-row tile, 256 threads = 8 waves) ----------
// wave w: row strip = w>>2 (16 rows), column tiles ct0, ct0+1 where ct0=(w&3)*2
template <int STRIDE, int KCH>
__device__ inline void gemm_hidden(const __bf16* fv, const __bf16* Wswz,
                                   const float* b1, float* hid, int tid) {
  int wave = tid >> 5, lane = tid & 31;
  int strip = wave >> 2;
  int ct0 = (wave & 3) * 2;
  v8f acc0 = {};
  v8f acc1 = {};
  for (int kc = 0; kc < KCH; ++kc) {
    v16bf a = load_a_bf16(fv + (strip * 16) * STRIDE + kc * 32, STRIDE, lane);
    v16bf b0 = load_b_swz(Wswz, kc, ct0, lane);
    v16bf b1v = load_b_swz(Wswz, kc, ct0 + 1, lane);
    acc0 = wmma_bf16(a, b0, acc0);
    acc1 = wmma_bf16(a, b1v, acc1);
  }
  int lo = lane & 15, hi = lane >> 4;
#pragma unroll
  for (int i = 0; i < 8; ++i) {
    int r = strip * 16 + hi * 8 + i;
    int c0 = ct0 * 16 + lo;
    int c1 = c0 + 16;
    hid[r * HID_STRIDE + c0] = acc0[i] + b1[c0];
    hid[r * HID_STRIDE + c1] = acc1[i] + b1[c1];
  }
}

// LayerNorm + leaky, 8 threads per row, result -> bf16 staging (WMMA-A friendly)
__device__ inline void ln_leaky_bf16(const float* hid, __bf16* hidb, const float* g,
                                     const float* be, int tid) {
  int row = tid >> 3, sub = tid & 7;
  const float* rp = hid + row * HID_STRIDE + sub * 16;
  float s = 0.f;
#pragma unroll
  for (int c = 0; c < 16; ++c) s += rp[c];
  float mu = grp8_sum(s) * (1.f / HDIM);
  float vs = 0.f;
#pragma unroll
  for (int c = 0; c < 16; ++c) {
    float d = rp[c] - mu;
    vs += d * d;
  }
  float rs = rsqrtf(grp8_sum(vs) * (1.f / HDIM) + 1e-5f);
  __bf16* op = hidb + row * HB_STRIDE + sub * 16;
#pragma unroll
  for (int c = 0; c < 16; ++c) {
    int col = sub * 16 + c;
    float x = (rp[c] - mu) * rs * g[col] + be[col];
    x = x >= 0.f ? x : 0.01f * x;
    op[c] = f2bf(x);
  }
}

// LayerNorm + leaky fused with the H->1 output head. sub==0 lane of each row
// group holds the full dot on return.
__device__ inline float ln_leaky_dot(const float* hid, const float* g, const float* be,
                                     const float* w2, int tid) {
  int row = tid >> 3, sub = tid & 7;
  const float* rp = hid + row * HID_STRIDE + sub * 16;
  float s = 0.f;
#pragma unroll
  for (int c = 0; c < 16; ++c) s += rp[c];
  float mu = grp8_sum(s) * (1.f / HDIM);
  float vs = 0.f;
#pragma unroll
  for (int c = 0; c < 16; ++c) {
    float d = rp[c] - mu;
    vs += d * d;
  }
  float rs = rsqrtf(grp8_sum(vs) * (1.f / HDIM) + 1e-5f);
  float dot = 0.f;
#pragma unroll
  for (int c = 0; c < 16; ++c) {
    int col = sub * 16 + c;
    float x = (rp[c] - mu) * rs * g[col] + be[col];
    x = x >= 0.f ? x : 0.01f * x;
    dot += x * w2[col];
  }
  return grp8_sum(dot);
}

// ---------- state / prep kernels ----------
__global__ void init_state_kernel(const float* v, const float* th, const float* pg,
                                  float* outv, float* outth, float* pgws, int N) {
  int i = blockIdx.x * 256 + threadIdx.x;
  if (i >= N) return;
  outv[i] = v[i];
  outth[i] = th[i];
  pgws[i] = pg[i];
}

// f32 [rows][128] -> bf16 fragment-swizzled [KC][8 ct][32 lane][16 j]
// element (kc,ct,lane,j) = W[kc*32 + (lane>>4)*16 + j][ct*16 + (lane&15)]
__global__ void convert_swz_kernel(const float* src, __bf16* dst, int rows, int KC) {
  int idx = blockIdx.x * 256 + threadIdx.x;
  if (idx >= KC * 4096) return;
  int j = idx & 15;
  int lane = (idx >> 4) & 31;
  int ct = (idx >> 9) & 7;
  int kc = idx >> 12;
  int row = kc * 32 + (lane >> 4) * 16 + j;
  int col = ct * 16 + (lane & 15);
  dst[idx] = f2bf(row < rows ? src[(size_t)row * HDIM + col] : 0.f);
}

// ---------- physics kernels ----------
__global__ void pjoule_kernel(const float* vA, const float* thA, const float* ea,
                              const int* src, const int* dst, const int* batch,
                              float* p_global, int E) {
  int e = blockIdx.x * 256 + threadIdx.x;
  if (e >= E) return;
  size_t b = (size_t)e * 7;
  float r = ea[b + 0], x = ea[b + 1], tau = ea[b + 5], shift = ea[b + 6];
  int si = src[e], di = dst[e];
  float vi = vA[si], vj = vA[di], ti = thA[si], tj = thA[di];
  float y_abs = rsqrtf(r * r + x * x);
  float delta = -atan2f(x, r);
  float t1 = -vi * vj * y_abs / tau *
             (cosf(ti - tj - delta - shift) + cosf(tj - ti - delta + shift));
  float vt = vi / tau;
  float cd = cosf(delta);
  float t2 = vt * vt * y_abs * cd;
  float t3 = vj * vj * y_abs * cd;
  atomicAdd(&p_global[batch[si]], fabsf(t1 + t2 + t3));
}

__global__ void pglobal_node_kernel(const float* vA, const float* pd, const float* shunt,
                                    const int* batch, float* p_global, int N) {
  int i = blockIdx.x * 256 + threadIdx.x;
  if (i >= N) return;
  float v = vA[i];
  atomicAdd(&p_global[batch[i]], pd[i] + v * v * shunt[(size_t)i * 2]);
}

__global__ void slack_kernel(const float* p_global, const float* gen, const float* lim,
                             const int* ptr, const int* sgidx, const int* sbidx,
                             float* pgA, int NG) {
  int g = blockIdx.x * 128 + threadIdx.x;
  if (g >= NG) return;
  float sum_all = 0.f;
  for (int i = ptr[g]; i < ptr[g + 1]; ++i) sum_all += gen[(size_t)i * 2];
  int sg = sgidx[g];
  float ps = gen[(size_t)sg * 2];
  float pmin = lim[(size_t)sg * 2];
  float pmax = lim[(size_t)sg * 2 + 1];
  float sum_ns = sum_all - ps;
  float pglob = p_global[g];
  float lamb = (pglob < sum_all)
                   ? (pglob - sum_ns - pmax) / (2.f * (ps - pmin))
                   : (pglob - sum_ns - 2.f * ps - pmax) / (2.f * (pmax - ps));
  lamb = fmaxf(lamb, 0.f);
  float pgs = (lamb < 0.5f) ? pmin + 2.f * (ps - pmin) * lamb
                            : 2.f * ps - pmax + 2.f * (pmax - ps) * lamb;
  pgA[sbidx[g]] = pgs;
}

__global__ void sflow_kernel(const float* vA, const float* thA, const float* ea,
                             const int* src, const int* dst, float* sfRe, float* sfIm,
                             int E) {
  int e = blockIdx.x * 256 + threadIdx.x;
  if (e >= E) return;
  size_t b = (size_t)e * 7;
  float r = ea[b + 0], x = ea[b + 1], bc = ea[b + 3], tau = ea[b + 5], shift = ea[b + 6];
  float den = r * r + x * x;
  float yRe = r / den, yIm = -x / den;
  float it = 1.f / tau, it2 = it * it;
  float YffRe = yRe * it2, YffIm = (yIm + 0.5f * bc) * it2;
  float cs, sn;
  sincosf(shift, &sn, &cs);
  float YftRe = -(yRe * cs - yIm * sn) * it;
  float YftIm = -(yRe * sn + yIm * cs) * it;
  float YtfRe = -(yRe * cs + yIm * sn) * it;
  float YtfIm = (yRe * sn - yIm * cs) * it;
  float YttRe = yRe, YttIm = yIm + 0.5f * bc;
  int si = src[e], di = dst[e];
  float vi = vA[si], vj = vA[di];
  float ci, sii, cj, sj;
  sincosf(thA[si], &sii, &ci);
  sincosf(thA[di], &sj, &cj);
  float VsRe = vi * ci, VsIm = vi * sii;
  float VdRe = vj * cj, VdIm = vj * sj;
  float IfRe = YffRe * VsRe - YffIm * VsIm + YftRe * VdRe - YftIm * VdIm;
  float IfIm = YffRe * VsIm + YffIm * VsRe + YftRe * VdIm + YftIm * VdRe;
  float SfRe = VsRe * IfRe + VsIm * IfIm;
  float SfIm = VsIm * IfRe - VsRe * IfIm;
  float ItRe = YtfRe * VsRe - YtfIm * VsIm + YttRe * VdRe - YttIm * VdIm;
  float ItIm = YtfRe * VsIm + YtfIm * VsRe + YttRe * VdIm + YttIm * VdRe;
  float StRe = VdRe * ItRe + VdIm * ItIm;
  float StIm = VdIm * ItRe - VdRe * ItIm;
  atomicAdd(&sfRe[si], SfRe);
  atomicAdd(&sfIm[si], SfIm);
  atomicAdd(&sfRe[di], StRe);
  atomicAdd(&sfIm[di], StIm);
}

__global__ __launch_bounds__(256) void balance_kernel(
    const float* vA, const float* pgA, float* qgA, const float* pd, const float* qd,
    const float* shunt, const float* sfRe, const float* sfIm, float* dpA, float* dqA,
    float* dssum, int slot, int writeQg, int N) {
  int i = blockIdx.x * 256 + threadIdx.x;
  int tid = threadIdx.x;
  float local = 0.f;
  if (i < N) {
    float v = vA[i], v2 = v * v;
    float gsh = shunt[(size_t)i * 2], bsh = shunt[(size_t)i * 2 + 1];
    float dp = pgA[i] - pd[i] - sfRe[i] - v2 * gsh;
    float base = -qd[i] - sfIm[i] + v2 * bsh;  // dq with qg = 0
    float qg;
    if (writeQg) {
      qg = -base;
      qgA[i] = qg;
    } else {
      qg = qgA[i];
    }
    float dq = qg + base;
    dpA[i] = dp;
    dqA[i] = dq;
    local = dp * dp + dq * dq;
  }
  __shared__ float red[256];
  red[tid] = local;
  __syncthreads();
  for (int s = 128; s > 0; s >>= 1) {
    if (tid < s) red[tid] += red[tid + s];
    __syncthreads();
  }
  if (tid == 0) atomicAdd(&dssum[slot], red[0]);
}

// ---------- message MLP over edges (fused scatter into agg) ----------
__global__ __launch_bounds__(256) void msg_mlp_kernel(
    const float* m, const float* ea, const int* src, const int* dst,
    const __bf16* W1s, const float* b1, const float* g1, const float* be1,
    const __bf16* W2s, const float* b2, float* agg, int E) {
  __shared__ __bf16 fv[32 * MSG_STRIDE];
  __shared__ float hid[32 * HID_STRIDE];
  __shared__ __bf16 hidb[32 * HB_STRIDE];
  __shared__ int sDst[32];
  int rowBase = blockIdx.x * 32;
  int tid = threadIdx.x;
  // fv fill: 42 float4 slots per row
  for (int idx = tid; idx < 32 * (MSG_STRIDE / 4); idx += 256) {
    int r = idx / (MSG_STRIDE / 4), q = idx - r * (MSG_STRIDE / 4);
    int e = rowBase + r;
    if (e >= E) e = E - 1;
    float v0 = 0.f, v1 = 0.f, v2 = 0.f, v3 = 0.f;
    if (q < 32) {
      const float4* mr = (const float4*)(m + (size_t)src[e] * HDIM);
      float4 t = mr[q];
      v0 = t.x; v1 = t.y; v2 = t.z; v3 = t.w;
    } else if (q == 32) {
      size_t b = (size_t)e * 7;
      v0 = ea[b + 0]; v1 = ea[b + 1]; v2 = ea[b + 3]; v3 = ea[b + 5];
    } else if (q == 33) {
      v0 = ea[(size_t)e * 7 + 6];
    }
    __bf16* fp = fv + r * MSG_STRIDE + q * 4;
    fp[0] = f2bf(v0); fp[1] = f2bf(v1); fp[2] = f2bf(v2); fp[3] = f2bf(v3);
  }
  if (tid < 32) {
    int e = rowBase + tid;
    sDst[tid] = dst[e < E ? e : E - 1];
  }
  __syncthreads();
  gemm_hidden<MSG_STRIDE, 5>(fv, W1s, b1, hid, tid);
  __syncthreads();
  ln_leaky_bf16(hid, hidb, g1, be1, tid);
  __syncthreads();
  int wave = tid >> 5, lane = tid & 31;
  int strip = wave >> 2, ct0 = (wave & 3) * 2;
  v8f acc0 = {};
  v8f acc1 = {};
  for (int kc = 0; kc < 4; ++kc) {
    v16bf a = load_a_bf16(hidb + (strip * 16) * HB_STRIDE + kc * 32, HB_STRIDE, lane);
    v16bf b0 = load_b_swz(W2s, kc, ct0, lane);
    v16bf b1v = load_b_swz(W2s, kc, ct0 + 1, lane);
    acc0 = wmma_bf16(a, b0, acc0);
    acc1 = wmma_bf16(a, b1v, acc1);
  }
  int lo = lane & 15, hi = lane >> 4;
#pragma unroll
  for (int i = 0; i < 8; ++i) {
    int r = strip * 16 + hi * 8 + i;
    int e = rowBase + r;
    if (e < E) {
      int d = sDst[r];
      int c0 = ct0 * 16 + lo, c1 = c0 + 16;
      atomicAdd(&agg[(size_t)d * HDIM + c0], acc0[i] + b2[c0]);
      atomicAdd(&agg[(size_t)d * HDIM + c1], acc1[i] + b2[c1]);
    }
  }
}

// ---------- three node MLPs (theta / v / m), fused over one fv tile ----------
__global__ __launch_bounds__(256) void node_mlp_kernel(
    float* vA, float* thA, const float* dp, const float* dq, float* m, const float* agg,
    const int* bus_type, const __bf16* ltW1, const float* ltB1, const float* ltG,
    const float* ltBe, const float* ltW2, const float* ltB2, const __bf16* lvW1,
    const float* lvB1, const float* lvG, const float* lvBe, const float* lvW2,
    const float* lvB2, const __bf16* lmW1, const float* lmB1, const float* lmG,
    const float* lmBe, const __bf16* lmW2, const float* lmB2, int N) {
  __shared__ __bf16 fv[32 * NODE_STRIDE];
  __shared__ float hid[32 * HID_STRIDE];
  __shared__ __bf16 hidb[32 * HB_STRIDE];
  int rowBase = blockIdx.x * 32;
  int tid = threadIdx.x;
  // fv fill: 74 float4 slots per row
  for (int idx = tid; idx < 32 * (NODE_STRIDE / 4); idx += 256) {
    int r = idx / (NODE_STRIDE / 4), q = idx - r * (NODE_STRIDE / 4);
    int node = rowBase + r;
    if (node >= N) node = N - 1;
    float v0 = 0.f, v1 = 0.f, v2 = 0.f, v3 = 0.f;
    if (q == 0) {
      v0 = vA[node]; v1 = thA[node]; v2 = dp[node]; v3 = dq[node];
    } else if (q < 33) {
      const float4* mr = (const float4*)(m + (size_t)node * HDIM);
      float4 t = mr[q - 1];
      v0 = t.x; v1 = t.y; v2 = t.z; v3 = t.w;
    } else if (q < 65) {
      const float4* ar = (const float4*)(agg + (size_t)node * HDIM);
      float4 t = ar[q - 33];
      v0 = t.x; v1 = t.y; v2 = t.z; v3 = t.w;
    }
    __bf16* fp = fv + r * NODE_STRIDE + q * 4;
    fp[0] = f2bf(v0); fp[1] = f2bf(v1); fp[2] = f2bf(v2); fp[3] = f2bf(v3);
  }
  __syncthreads();

  // theta update
  gemm_hidden<NODE_STRIDE, 9>(fv, ltW1, ltB1, hid, tid);
  __syncthreads();
  {
    float dot = ln_leaky_dot(hid, ltG, ltBe, ltW2, tid);
    if ((tid & 7) == 0) {
      int node = rowBase + (tid >> 3);
      if (node < N) thA[node] += dot + ltB2[0];
    }
  }
  __syncthreads();

  // v update (masked for generator buses)
  gemm_hidden<NODE_STRIDE, 9>(fv, lvW1, lvB1, hid, tid);
  __syncthreads();
  {
    float dot = ln_leaky_dot(hid, lvG, lvBe, lvW2, tid);
    if ((tid & 7) == 0) {
      int node = rowBase + (tid >> 3);
      if (node < N && bus_type[node] != 2) vA[node] += dot + lvB2[0];
    }
  }
  __syncthreads();

  // m update (second WMMA GEMM, residual via C operand)
  gemm_hidden<NODE_STRIDE, 9>(fv, lmW1, lmB1, hid, tid);
  __syncthreads();
  ln_leaky_bf16(hid, hidb, lmG, lmBe, tid);
  __syncthreads();
  {
    int wave = tid >> 5, lane = tid & 31;
    int strip = wave >> 2, ct0 = (wave & 3) * 2;
    int lo = lane & 15, hi = lane >> 4;
    v8f acc0 = {};
    v8f acc1 = {};
#pragma unroll
    for (int i = 0; i < 8; ++i) {
      int r = strip * 16 + hi * 8 + i;
      int node = rowBase + r;
      if (node >= N) node = N - 1;
      acc0[i] = m[(size_t)node * HDIM + ct0 * 16 + lo];
      acc1[i] = m[(size_t)node * HDIM + ct0 * 16 + 16 + lo];
    }
    for (int kc = 0; kc < 4; ++kc) {
      v16bf a = load_a_bf16(hidb + (strip * 16) * HB_STRIDE + kc * 32, HB_STRIDE, lane);
      v16bf b0 = load_b_swz(lmW2, kc, ct0, lane);
      v16bf b1v = load_b_swz(lmW2, kc, ct0 + 1, lane);
      acc0 = wmma_bf16(a, b0, acc0);
      acc1 = wmma_bf16(a, b1v, acc1);
    }
#pragma unroll
    for (int i = 0; i < 8; ++i) {
      int r = strip * 16 + hi * 8 + i;
      int node = rowBase + r;
      if (node < N) {
        int c0 = ct0 * 16 + lo, c1 = c0 + 16;
        m[(size_t)node * HDIM + c0] = acc0[i] + lmB2[c0];
        m[(size_t)node * HDIM + c1] = acc1[i] + lmB2[c1];
      }
    }
  }
}

__global__ void finalize_kernel(const float* dssum, float* out, int N) {
  if (threadIdx.x == 0 && blockIdx.x == 0) {
    float invN = 1.0f / (float)N;
    float last = dssum[KITER] * invN;
    float total = last;
    float w = GAMMA_F;
    for (int k = KITER - 1; k >= 0; --k) {
      total += dssum[k] * invN * w;
      w *= GAMMA_F;
    }
    out[2 * N] = last;
    out[2 * N + 1] = total;
  }
}

// ---------- host driver ----------
extern "C" void kernel_launch(void* const* d_in, const int* in_sizes, int n_in,
                              void* d_out, int out_size, void* d_ws, size_t ws_size,
                              hipStream_t stream) {
  (void)n_in;
  (void)out_size;
  (void)ws_size;
  const float* in_v = (const float*)d_in[0];
  const float* in_th = (const float*)d_in[1];
  const float* in_pd = (const float*)d_in[2];
  const float* in_qd = (const float*)d_in[3];
  const float* in_pg = (const float*)d_in[4];
  const float* in_shunt = (const float*)d_in[6];
  const float* in_ea = (const float*)d_in[7];
  const float* in_gen = (const float*)d_in[8];
  const float* in_lim = (const float*)d_in[9];
  const float* phi_w1 = (const float*)d_in[10];
  const float* phi_b1 = (const float*)d_in[11];
  const float* phi_g = (const float*)d_in[12];
  const float* phi_be = (const float*)d_in[13];
  const float* phi_w2 = (const float*)d_in[14];
  const float* phi_b2 = (const float*)d_in[15];
  const float* lt_w1 = (const float*)d_in[16];
  const float* lt_b1 = (const float*)d_in[17];
  const float* lt_g = (const float*)d_in[18];
  const float* lt_be = (const float*)d_in[19];
  const float* lt_w2 = (const float*)d_in[20];
  const float* lt_b2 = (const float*)d_in[21];
  const float* lv_w1 = (const float*)d_in[22];
  const float* lv_b1 = (const float*)d_in[23];
  const float* lv_g = (const float*)d_in[24];
  const float* lv_be = (const float*)d_in[25];
  const float* lv_w2 = (const float*)d_in[26];
  const float* lv_b2 = (const float*)d_in[27];
  const float* lm_w1 = (const float*)d_in[28];
  const float* lm_b1 = (const float*)d_in[29];
  const float* lm_g = (const float*)d_in[30];
  const float* lm_be = (const float*)d_in[31];
  const float* lm_w2 = (const float*)d_in[32];
  const float* lm_b2 = (const float*)d_in[33];
  const int* edge_index = (const int*)d_in[34];
  const int* bus_type = (const int*)d_in[35];
  const int* batch = (const int*)d_in[36];
  const int* gptr = (const int*)d_in[37];
  const int* sgidx = (const int*)d_in[38];
  const int* sbidx = (const int*)d_in[39];

  const int N = in_sizes[0];
  const int E = in_sizes[34] / 2;
  const int NG = in_sizes[37] - 1;
  const int* srcA = edge_index;
  const int* dstA = edge_index + E;

  float* out = (float*)d_out;
  float* outv = out;
  float* outth = out + N;

  char* ws = (char*)d_ws;
  size_t off = 0;
  auto carve = [&](size_t bytes) -> char* {
    char* p = ws + off;
    off = (off + bytes + 255) & ~(size_t)255;
    return p;
  };
  float* pgws = (float*)carve((size_t)N * 4);
  float* qgws = (float*)carve((size_t)N * 4);
  float* dpws = (float*)carve((size_t)N * 4);
  float* dqws = (float*)carve((size_t)N * 4);
  float* sfRe = (float*)carve((size_t)N * 4);
  float* sfIm = (float*)carve((size_t)N * 4);
  float* p_global = (float*)carve((size_t)NG * 4);
  float* dssum = (float*)carve(64);
  float* mbuf = (float*)carve((size_t)N * HDIM * 4);
  float* aggbuf = (float*)carve((size_t)N * HDIM * 4);
  __bf16* phiW1p = (__bf16*)carve((size_t)160 * HDIM * 2);
  __bf16* phiW2p = (__bf16*)carve((size_t)HDIM * HDIM * 2);
  __bf16* ltW1p = (__bf16*)carve((size_t)KITER * 288 * HDIM * 2);
  __bf16* lvW1p = (__bf16*)carve((size_t)KITER * 288 * HDIM * 2);
  __bf16* lmW1p = (__bf16*)carve((size_t)KITER * 288 * HDIM * 2);
  __bf16* lmW2p = (__bf16*)carve((size_t)KITER * HDIM * HDIM * 2);

  int nb = (N + 255) / 256;
  int eb = (E + 255) / 256;

  hipMemsetAsync(mbuf, 0, (size_t)N * HDIM * 4, stream);
  hipMemsetAsync(dssum, 0, 64, stream);
  init_state_kernel<<<nb, 256, 0, stream>>>(in_v, in_th, in_pg, outv, outth, pgws, N);

  auto convLaunch = [&](const float* s, __bf16* d, int rows, int KC) {
    int tot = KC * 4096;
    convert_swz_kernel<<<(tot + 255) / 256, 256, 0, stream>>>(s, d, rows, KC);
  };
  convLaunch(phi_w1, phiW1p, HDIM + 5, 5);
  convLaunch(phi_w2, phiW2p, HDIM, 4);
  for (int k = 0; k < KITER; ++k) {
    convLaunch(lt_w1 + (size_t)k * 260 * HDIM, ltW1p + (size_t)k * 288 * HDIM, 260, 9);
    convLaunch(lv_w1 + (size_t)k * 260 * HDIM, lvW1p + (size_t)k * 288 * HDIM, 260, 9);
    convLaunch(lm_w1 + (size_t)k * 260 * HDIM, lmW1p + (size_t)k * 288 * HDIM, 260, 9);
    convLaunch(lm_w2 + (size_t)k * HDIM * HDIM, lmW2p + (size_t)k * HDIM * HDIM, HDIM,
               4);
  }

  for (int k = 0; k < KITER; ++k) {
    hipMemsetAsync(p_global, 0, (size_t)NG * 4, stream);
    hipMemsetAsync(sfRe, 0, (size_t)N * 4, stream);
    hipMemsetAsync(sfIm, 0, (size_t)N * 4, stream);
    hipMemsetAsync(aggbuf, 0, (size_t)N * HDIM * 4, stream);

    pjoule_kernel<<<eb, 256, 0, stream>>>(outv, outth, in_ea, srcA, dstA, batch,
                                          p_global, E);
    pglobal_node_kernel<<<nb, 256, 0, stream>>>(outv, in_pd, in_shunt, batch, p_global,
                                                N);
    slack_kernel<<<(NG + 127) / 128, 128, 0, stream>>>(p_global, in_gen, in_lim, gptr,
                                                       sgidx, sbidx, pgws, NG);
    sflow_kernel<<<eb, 256, 0, stream>>>(outv, outth, in_ea, srcA, dstA, sfRe, sfIm, E);
    balance_kernel<<<nb, 256, 0, stream>>>(outv, pgws, qgws, in_pd, in_qd, in_shunt,
                                           sfRe, sfIm, dpws, dqws, dssum, k, 1, N);
    msg_mlp_kernel<<<(E + 31) / 32, 256, 0, stream>>>(mbuf, in_ea, srcA, dstA, phiW1p,
                                                      phi_b1, phi_g, phi_be, phiW2p,
                                                      phi_b2, aggbuf, E);
    node_mlp_kernel<<<(N + 31) / 32, 256, 0, stream>>>(
        outv, outth, dpws, dqws, mbuf, aggbuf, bus_type,
        ltW1p + (size_t)k * 288 * HDIM, lt_b1 + (size_t)k * HDIM,
        lt_g + (size_t)k * HDIM, lt_be + (size_t)k * HDIM, lt_w2 + (size_t)k * HDIM,
        lt_b2 + k, lvW1p + (size_t)k * 288 * HDIM, lv_b1 + (size_t)k * HDIM,
        lv_g + (size_t)k * HDIM, lv_be + (size_t)k * HDIM, lv_w2 + (size_t)k * HDIM,
        lv_b2 + k, lmW1p + (size_t)k * 288 * HDIM, lm_b1 + (size_t)k * HDIM,
        lm_g + (size_t)k * HDIM, lm_be + (size_t)k * HDIM,
        lmW2p + (size_t)k * HDIM * HDIM, lm_b2 + (size_t)k * HDIM, N);
  }

  hipMemsetAsync(sfRe, 0, (size_t)N * 4, stream);
  hipMemsetAsync(sfIm, 0, (size_t)N * 4, stream);
  sflow_kernel<<<eb, 256, 0, stream>>>(outv, outth, in_ea, srcA, dstA, sfRe, sfIm, E);
  balance_kernel<<<nb, 256, 0, stream>>>(outv, pgws, qgws, in_pd, in_qd, in_shunt, sfRe,
                                         sfIm, dpws, dqws, dssum, KITER, 0, N);
  finalize_kernel<<<1, 32, 0, stream>>>(dssum, out, N);
}
